// EdgeConvolutionLayer_5677946765707
// MI455X (gfx1250) — matrix operand order repaired and checked by reading, not compiled, not run on hardware
//
#include <hip/hip_runtime.h>

// ---------------------------------------------------------------------------
// EdgeConv (DGCNN) for MI455X / gfx1250, wave32, WMMA f16 16x16x32.
//
// B=8, C=64, N=4096, O=64, K=20.
//
// Algebraic restructure:
//   h[b,o,n,k] = W1.(x_nbr - x_ctr) + W2.x_ctr  (+bias, which cancels in BN)
//              = P[b,o,idx[b,n,k]] + Q[b,o,n]
//   with P = W1 @ X,  Q = (W2 - W1) @ X   (tiny GEMMs, WMMA)
// knn: fused  S = X^T X  (WMMA, f16 in / f32 acc)  + per-row top-20 in regs,
//      with global_prefetch of the next column tile hidden under top-K VALU.
// ---------------------------------------------------------------------------

#define NPTS 4096
#define CCH  64
#define TOPK 20
#define BATCH 8
#define OCH  64

typedef __attribute__((ext_vector_type(16))) _Float16    v16h;
typedef __attribute__((ext_vector_type(8)))  float       v8f;
typedef __attribute__((ext_vector_type(8)))  unsigned int v8u;

// ---- A fragment (16x32 f16, MxK), contiguous k order (same as load_fragB) --------------------
// lane 0-15: M=lane, VGPR v: K=2v+16h  (h = lane>>4)
// Source tile is m-major with K contiguous -> each VGPR = one b32 load.
__device__ __forceinline__ v16h load_fragA(const _Float16* tile_m_major,
                                           int m, int pitch_h, int half,
                                           int kbase) {
  const unsigned int* rowp = (const unsigned int*)(tile_m_major + m * pitch_h);
  v8u u;
#pragma unroll
  for (int v = 0; v < 8; ++v) {
    int k = kbase + 2 * v + 16 * half;
    u[v] = rowp[k >> 1];
  }
  return __builtin_bit_cast(v16h, u);
}

// ---- B fragment (32x16 f16, KxN) ---------------------------------------
// lane 0-15: N=lane, K=0..15 (V v: K=2v,2v+1); lanes 16-31: K=16..31.
// Source tile is n-major with K contiguous.
__device__ __forceinline__ v16h load_fragB(const _Float16* tile_n_major,
                                           int n, int pitch_h, int half,
                                           int kbase) {
  const unsigned int* colp = (const unsigned int*)(tile_n_major + n * pitch_h);
  int kb0 = kbase + 16 * half;
  v8u u;
#pragma unroll
  for (int v = 0; v < 8; ++v) u[v] = colp[(kb0 >> 1) + v];
  return __builtin_bit_cast(v16h, u);
}

// ---------------------------------------------------------------------------
// Kernel 0a: x f32 -> f16, and xx[b,n] = sum_c x^2
// ---------------------------------------------------------------------------
__global__ void prep_x_kernel(const float* __restrict__ x,
                              _Float16* __restrict__ xh,
                              float* __restrict__ xx) {
  int gid = blockIdx.x * blockDim.x + threadIdx.x;      // 0 .. B*N-1
  int b = gid >> 12;
  int n = gid & (NPTS - 1);
  const float* xp = x + (size_t)b * CCH * NPTS + n;
  _Float16*    hp = xh + (size_t)b * CCH * NPTS + n;
  float s = 0.f;
#pragma unroll
  for (int c = 0; c < CCH; ++c) {
    float v = xp[(size_t)c * NPTS];
    s += v * v;
    hp[(size_t)c * NPTS] = (_Float16)v;
  }
  xx[gid] = s;
}

// ---------------------------------------------------------------------------
// Kernel 0b: build Wp[128][64] f16 : rows 0-63 = W1, rows 64-127 = W2 - W1
// (conv bias is dropped: it cancels exactly in (h - mean) of BatchNorm)
// ---------------------------------------------------------------------------
__global__ void prep_w_kernel(const float* __restrict__ W,
                              _Float16* __restrict__ wp) {
  for (int linear = threadIdx.x; linear < 128 * CCH; linear += blockDim.x) {
    int r = linear >> 6, c = linear & 63;
    float v = (r < OCH) ? W[r * (2 * CCH) + c]
                        : (W[(r - OCH) * (2 * CCH) + CCH + c] -
                           W[(r - OCH) * (2 * CCH) + c]);
    wp[r * CCH + c] = (_Float16)v;
  }
}

// ---------------------------------------------------------------------------
// Kernel 1: fused  S = X_rows^T X_cols  (WMMA)  +  per-row top-20.
// Block: 128 threads (4 waves), owns 128 rows of one batch.
// Col tiles of 64; S tile staged in LDS; thread t keeps sorted top-20 in regs.
// Next tile is global_prefetch'ed at the start of the top-K phase.
// ---------------------------------------------------------------------------
__global__ void __launch_bounds__(128)
knn_topk_kernel(const _Float16* __restrict__ xh,
                const float* __restrict__ xx,
                int* __restrict__ knn_idx) {
  __shared__ _Float16 xrt[128 * 68];   // rows tile, m-major (K contiguous)
  __shared__ _Float16 xct[64 * 68];    // cols tile, n-major
  __shared__ float    S[128 * 68];     // inner-product tile
  __shared__ float    xxc[64];

  const int tid  = threadIdx.x;
  const int b    = blockIdx.y;
  const int row0 = blockIdx.x * 128;
  const int lane = tid & 31;
  const int wv   = tid >> 5;           // 0..3
  const int hf   = (lane >> 4) & 1;
  const int l15  = lane & 15;

  const _Float16* xb = xh + (size_t)b * CCH * NPTS;

  for (int linear = tid; linear < 128 * CCH; linear += 128) {
    int i = linear & 127;
    int c = linear >> 7;
    xrt[i * 68 + c] = xb[(size_t)c * NPTS + row0 + i];
  }
  __syncthreads();

  // Resident A fragments: row tiles ti = 2*wv+s, K blocks kb = 0,1
  v16h fragA[2][2];
#pragma unroll
  for (int s = 0; s < 2; ++s)
#pragma unroll
    for (int kb = 0; kb < 2; ++kb)
      fragA[s][kb] = load_fragA(xrt, (2 * wv + s) * 16 + l15, 68, hf, kb * 32);

  float tv[TOPK];
  int   tix[TOPK];
#pragma unroll
  for (int p = 0; p < TOPK; ++p) { tv[p] = -3.0e38f; tix[p] = 0; }
  const float xxr = xx[b * NPTS + row0 + tid];

  for (int col0 = 0; col0 < NPTS; col0 += 64) {
    for (int linear = tid; linear < 64 * CCH; linear += 128) {
      int j = linear & 63;
      int c = linear >> 6;
      xct[j * 68 + c] = xb[(size_t)c * NPTS + col0 + j];
    }
    if (tid < 64) xxc[tid] = xx[b * NPTS + col0 + tid];
    __syncthreads();

#pragma unroll
    for (int tj = 0; tj < 4; ++tj) {
      v16h fragB0 = load_fragB(xct, tj * 16 + l15, 68, hf, 0);
      v16h fragB1 = load_fragB(xct, tj * 16 + l15, 68, hf, 32);
#pragma unroll
      for (int s = 0; s < 2; ++s) {
        v8f acc = {};
        acc = __builtin_amdgcn_wmma_f32_16x16x32_f16(
            false, fragA[s][0], false, fragB0, (short)0, acc, false, false);
        acc = __builtin_amdgcn_wmma_f32_16x16x32_f16(
            false, fragA[s][1], false, fragB1, (short)0, acc, false, false);
        int ti = 2 * wv + s;
#pragma unroll
        for (int r = 0; r < 8; ++r)
          S[(ti * 16 + r + 8 * hf) * 68 + tj * 16 + l15] = acc[r];
      }
    }
    __syncthreads();

    // Prefetch next column tile into WGP$/L2 while the top-K VALU phase runs.
    // __builtin_prefetch lowers to global_prefetch_b8 on gfx1250 (one
    // cacheline per enabled lane): 128 lanes cover the 64ch x 64pt f16 tile.
    if (col0 + 64 < NPTS) {
      int pc = tid >> 1;                    // channel 0..63
      int pj = (tid & 1) * 32;              // two 64B chunks per channel row
      __builtin_prefetch(&xb[(size_t)pc * NPTS + col0 + 64 + pj], 0, 2);
      if (tid < 2) __builtin_prefetch(&xx[b * NPTS + col0 + 64 + tid * 32], 0, 2);
    }

    // top-20 update: thread t owns row t. dist = 2*S - ||xr|| - ||xc||
    const float* srow = S + tid * 68;
    for (int j = 0; j < 64; ++j) {
      float d = 2.0f * srow[j] - xxr - xxc[j];
      if (d > tv[TOPK - 1]) {
        tv[TOPK - 1] = d;
        tix[TOPK - 1] = col0 + j;
#pragma unroll
        for (int p = TOPK - 1; p > 0; --p) {
          if (tv[p] > tv[p - 1]) {
            float fv = tv[p]; tv[p] = tv[p - 1]; tv[p - 1] = fv;
            int   iv = tix[p]; tix[p] = tix[p - 1]; tix[p - 1] = iv;
          }
        }
      }
    }
    __syncthreads();
  }

  int* op = knn_idx + ((size_t)b * NPTS + row0 + tid) * TOPK;
#pragma unroll
  for (int p = 0; p < TOPK; ++p) op[p] = tix[p];
}

// ---------------------------------------------------------------------------
// Kernel 2: [P;Q] = Wp (128x64) @ X (64xN) per batch, WMMA.
// Block: 256 threads (8 waves), N tile 128. Wave w owns row tile ti = w.
// ---------------------------------------------------------------------------
__global__ void __launch_bounds__(256)
proj_kernel(const _Float16* __restrict__ xh,
            const _Float16* __restrict__ wp,
            float* __restrict__ P,
            float* __restrict__ Qm) {
  __shared__ _Float16 xct[128 * 68];   // n-major, K contiguous

  const int tid  = threadIdx.x;
  const int b    = blockIdx.y;
  const int col0 = blockIdx.x * 128;
  const int lane = tid & 31;
  const int wv   = tid >> 5;           // 0..7
  const int hf   = (lane >> 4) & 1;
  const int l15  = lane & 15;

  const _Float16* xb = xh + (size_t)b * CCH * NPTS;
  for (int linear = tid; linear < 128 * CCH; linear += 256) {
    int i = linear & 127;
    int c = linear >> 7;
    xct[i * 68 + c] = xb[(size_t)c * NPTS + col0 + i];
  }
  __syncthreads();

  // A fragments from Wp (global, tiny + L2-hot), m-major pitch 64
  v16h fragA0 = load_fragA(wp, wv * 16 + l15, CCH, hf, 0);
  v16h fragA1 = load_fragA(wp, wv * 16 + l15, CCH, hf, 32);

  float* outbase = (wv < 4) ? P : Qm;
  int    obase   = (wv < 4) ? (wv * 16 + 8 * hf) : (wv * 16 + 8 * hf - OCH);

#pragma unroll
  for (int tj = 0; tj < 8; ++tj) {
    v16h fragB0 = load_fragB(xct, tj * 16 + l15, 68, hf, 0);
    v16h fragB1 = load_fragB(xct, tj * 16 + l15, 68, hf, 32);
    v8f acc = {};
    acc = __builtin_amdgcn_wmma_f32_16x16x32_f16(
        false, fragA0, false, fragB0, (short)0, acc, false, false);
    acc = __builtin_amdgcn_wmma_f32_16x16x32_f16(
        false, fragA1, false, fragB1, (short)0, acc, false, false);
#pragma unroll
    for (int r = 0; r < 8; ++r)
      outbase[((size_t)b * OCH + obase + r) * NPTS + col0 + tj * 16 + l15] = acc[r];
  }
}

// ---------------------------------------------------------------------------
// Kernel 3a/3b: BatchNorm stats (sum, sumsq per channel) via atomics.
// ---------------------------------------------------------------------------
__global__ void zero_stats_kernel(float* __restrict__ stats) {
  if (threadIdx.x < 2 * OCH) stats[threadIdx.x] = 0.f;
}

__global__ void __launch_bounds__(256)
bn_stats_kernel(const float* __restrict__ P,
                const float* __restrict__ Qm,
                const int* __restrict__ knn_idx,
                float* __restrict__ stats) {
  __shared__ int   idxl[128 * TOPK];
  __shared__ float reds[256];
  __shared__ float redq[256];

  const int tid = threadIdx.x;
  const int b   = blockIdx.y;
  const int n0  = blockIdx.x * 128;

  for (int linear = tid; linear < 128 * TOPK; linear += 256)
    idxl[linear] = knn_idx[((size_t)b * NPTS + n0) * TOPK + linear];
  __syncthreads();

  const int o = tid & 63;
  const int g = tid >> 6;
  const float* Pb = P  + ((size_t)b * OCH + o) * NPTS;
  const float* Qb = Qm + ((size_t)b * OCH + o) * NPTS;

  float s = 0.f, s2 = 0.f;
  for (int nl = g; nl < 128; nl += 4) {
    float q = Qb[n0 + nl];
#pragma unroll
    for (int k = 0; k < TOPK; ++k) {
      float h = Pb[idxl[nl * TOPK + k]] + q;
      s  += h;
      s2 += h * h;
    }
  }
  reds[tid] = s;
  redq[tid] = s2;
  __syncthreads();
  if (tid < 64) {
    s  = reds[tid] + reds[tid + 64] + reds[tid + 128] + reds[tid + 192];
    s2 = redq[tid] + redq[tid + 64] + redq[tid + 128] + redq[tid + 192];
    atomicAdd(&stats[o], s);
    atomicAdd(&stats[OCH + o], s2);
  }
}

// ---------------------------------------------------------------------------
// Kernel 4: y = max_k leakyrelu( gamma*(h-mean)*rsqrt(var+eps)+beta )
// ---------------------------------------------------------------------------
__global__ void __launch_bounds__(256)
finalize_kernel(const float* __restrict__ P,
                const float* __restrict__ Qm,
                const int* __restrict__ knn_idx,
                const float* __restrict__ stats,
                const float* __restrict__ gamma,
                const float* __restrict__ beta,
                float* __restrict__ out) {
  __shared__ int idxl[128 * TOPK];

  const int tid = threadIdx.x;
  const int b   = blockIdx.y;
  const int n0  = blockIdx.x * 128;

  for (int linear = tid; linear < 128 * TOPK; linear += 256)
    idxl[linear] = knn_idx[((size_t)b * NPTS + n0) * TOPK + linear];
  __syncthreads();

  const int o = tid & 63;
  const int g = tid >> 6;

  const float cntinv = 1.0f / ((float)BATCH * (float)NPTS * (float)TOPK);
  float mean = stats[o] * cntinv;
  float var  = stats[OCH + o] * cntinv - mean * mean;
  float sc   = gamma[o] * rsqrtf(var + 1e-5f);
  float sh   = beta[o] - mean * sc;

  const float* Pb = P  + ((size_t)b * OCH + o) * NPTS;
  const float* Qb = Qm + ((size_t)b * OCH + o) * NPTS;
  float* ob = out + ((size_t)b * OCH + o) * NPTS;

  for (int nl = g; nl < 128; nl += 4) {
    float q = Qb[n0 + nl];
    float m = -3.0e38f;
#pragma unroll
    for (int k = 0; k < TOPK; ++k) {
      float h = Pb[idxl[nl * TOPK + k]] + q;
      float y = fmaf(sc, h, sh);
      y = (y > 0.f) ? y : 0.2f * y;
      m = fmaxf(m, y);
    }
    ob[n0 + nl] = m;
  }
}

// ---------------------------------------------------------------------------
// Launch
// ---------------------------------------------------------------------------
extern "C" void kernel_launch(void* const* d_in, const int* in_sizes, int n_in,
                              void* d_out, int out_size, void* d_ws, size_t ws_size,
                              hipStream_t stream) {
  (void)in_sizes; (void)n_in; (void)out_size; (void)ws_size;

  const float* x     = (const float*)d_in[0];   // [B,C,N]
  const float* W     = (const float*)d_in[1];   // [O,2C]
  // d_in[2] = conv bias: cancels exactly in BatchNorm's (h - mean); unused.
  const float* gamma = (const float*)d_in[3];
  const float* beta  = (const float*)d_in[4];
  float* out = (float*)d_out;                   // [B,O,N]

  char* ws = (char*)d_ws;
  const size_t OFF_XH    = 0;                                // f16 x     4 MB
  const size_t OFF_WP    = OFF_XH + (size_t)BATCH * CCH * NPTS * 2;  // 16 KB
  const size_t OFF_XX    = OFF_WP + 128 * CCH * 2 + 256;
  const size_t OFF_IDX   = OFF_XX + (size_t)BATCH * NPTS * 4 + 256;
  const size_t OFF_P     = OFF_IDX + (size_t)BATCH * NPTS * TOPK * 4 + 256;
  const size_t OFF_Q     = OFF_P + (size_t)BATCH * OCH * NPTS * 4 + 256;
  const size_t OFF_STATS = OFF_Q + (size_t)BATCH * OCH * NPTS * 4 + 256;

  _Float16* xh    = (_Float16*)(ws + OFF_XH);
  _Float16* wp    = (_Float16*)(ws + OFF_WP);
  float*    xx    = (float*)(ws + OFF_XX);
  int*      idx   = (int*)(ws + OFF_IDX);
  float*    P     = (float*)(ws + OFF_P);
  float*    Qm    = (float*)(ws + OFF_Q);
  float*    stats = (float*)(ws + OFF_STATS);

  prep_x_kernel<<<(BATCH * NPTS) / 256, 256, 0, stream>>>(x, xh, xx);
  prep_w_kernel<<<1, 256, 0, stream>>>(W, wp);

  knn_topk_kernel<<<dim3(NPTS / 128, BATCH), 128, 0, stream>>>(xh, xx, idx);
  proj_kernel<<<dim3(NPTS / 128, BATCH), 256, 0, stream>>>(xh, wp, P, Qm);

  zero_stats_kernel<<<1, 128, 0, stream>>>(stats);
  bn_stats_kernel<<<dim3(NPTS / 128, BATCH), 256, 0, stream>>>(P, Qm, idx, stats);
  finalize_kernel<<<dim3(NPTS / 128, BATCH), 256, 0, stream>>>(P, Qm, idx, stats,
                                                               gamma, beta, out);
}
